// TGCN_49374944035412
// MI455X (gfx1250) — matrix operand order repaired and checked by reading, not compiled, not run on hardware
//
#include <hip/hip_runtime.h>
#include <hip/hip_bf16.h>
#include <math.h>

// ---------------- types ----------------
typedef __attribute__((ext_vector_type(16))) __bf16 v16bf;
typedef __attribute__((ext_vector_type(8)))  __bf16 v8bf;
typedef __attribute__((ext_vector_type(8)))  float  v8f;
typedef int v4i __attribute__((vector_size(16)));   // matches async-builtin pointee

#define Bb   32
#define Ss   48
#define Nn   1000
#define NP   1024      // padded node dim (K of aggregation GEMM)
#define Hh   64
#define FCAT 104       // 40 feat + 64 hidden
#define FP   112       // padded to 7 wmma tiles of 16
#define G2   128       // 2*H gate channels
#define LDSP 136       // padded LDS row stride for AG scratch (halves)
#define KC   64        // K-chunk staged in LDS
#define BST  72        // LDS B-chunk row stride (halves): 64 + 8 pad (banks (f*36)%64 distinct)
#define NKC  (NP / KC) // 16 chunks

#if __has_builtin(__builtin_amdgcn_global_load_async_to_lds_b128) && \
    __has_builtin(__builtin_amdgcn_s_wait_asynccnt)
#define HAVE_ASYNC 1
#else
#define HAVE_ASYNC 0
#endif

static __device__ __forceinline__ v16bf cat16(v8bf lo, v8bf hi) {
  return __builtin_shufflevector(lo, hi, 0,1,2,3,4,5,6,7,8,9,10,11,12,13,14,15);
}
static __device__ __forceinline__ v8f zero8() {
  v8f z;
  #pragma unroll
  for (int i = 0; i < 8; ++i) z[i] = 0.0f;
  return z;
}

// copy one K-chunk (112 rows x 64 halves) of cat_b into LDS, 7 B128 segments/thread
static __device__ __forceinline__ void stage_chunk(const __bf16* __restrict__ catB,
                                                   __bf16* buf, int kc, int tid) {
  #pragma unroll
  for (int i = 0; i < 7; ++i) {
    int s = i * 128 + tid;            // 0..895
    int row = s >> 3, seg = s & 7;    // row 0..111, seg 0..7 (8 halves each)
    const __bf16* g = catB + (size_t)row * NP + kc * KC + seg * 8;
    __bf16* l = buf + row * BST + seg * 8;
#if HAVE_ASYNC
    __builtin_amdgcn_global_load_async_to_lds_b128(
        (__attribute__((address_space(1))) v4i*)g,
        (__attribute__((address_space(3))) v4i*)l, 0, 0);
#else
    *(v8bf*)l = *(const v8bf*)g;
#endif
  }
}
static __device__ __forceinline__ void wait_async_outstanding7() {
#if HAVE_ASYNC
  __builtin_amdgcn_s_wait_asynccnt(7);
#endif
}
static __device__ __forceinline__ void wait_async_all() {
#if HAVE_ASYNC
  __builtin_amdgcn_s_wait_asynccnt(0);
#endif
}

// ---------------- setup kernels ----------------
__global__ void k_dinv(const float* __restrict__ adj, float* __restrict__ dinv) {
  int n = blockIdx.x * blockDim.x + threadIdx.x;
  if (n >= Nn) return;
  float s = 1.0f; // self loop
  const float* row = adj + (size_t)n * Nn;
  for (int j = 0; j < Nn; ++j) s += row[j];
  dinv[n] = rsqrtf(s);
}

__global__ void k_buildL(const float* __restrict__ adj, const float* __restrict__ dinv,
                         __bf16* __restrict__ Lbf) {
  int idx = blockIdx.x * blockDim.x + threadIdx.x;  // < NP*NP
  int m = idx >> 10, n = idx & (NP - 1);
  float v = 0.0f;
  if (m < Nn && n < Nn) {
    float a = adj[(size_t)m * Nn + n] + (m == n ? 1.0f : 0.0f);
    v = dinv[m] * a * dinv[n];
  }
  Lbf[idx] = (__bf16)v;
}

__global__ void k_buildW(const float* __restrict__ W1, const float* __restrict__ W2,
                         __bf16* __restrict__ W1T, __bf16* __restrict__ W2T) {
  int idx = blockIdx.x * blockDim.x + threadIdx.x;
  if (idx < G2 * 128) {               // W1T[out=128][k=128]
    int o = idx >> 7, k = idx & 127;
    W1T[idx] = (__bf16)((k < FCAT) ? W1[(size_t)k * G2 + o] : 0.0f);
  } else if (idx < G2 * 128 + Hh * 128) {  // W2T[out=64][k=128]
    int i2 = idx - G2 * 128;
    int o = i2 >> 7, k = i2 & 127;
    W2T[i2] = (__bf16)((k < FCAT) ? W2[(size_t)k * Hh + o] : 0.0f);
  }
}

// ---------------- per-step: build transposed cat (bf16) ----------------
// catT layout: [b][f(0..111)][n(0..1023)]
__global__ void k_cat1(const float* __restrict__ inp, int t,
                       const float* __restrict__ Wo, const float* __restrict__ bo,
                       const float* __restrict__ Wd, const float* __restrict__ bd,
                       const float* __restrict__ We, const float* __restrict__ be,
                       const float* __restrict__ Wi, const float* __restrict__ bi,
                       const float* __restrict__ h, __bf16* __restrict__ catT) {
  int idx = blockIdx.x * blockDim.x + threadIdx.x;   // Bb*NP threads
  int b = idx >> 10, n = idx & (NP - 1);
  __bf16* out = catT + ((size_t)b * FP) * NP + n;
  if (n >= Nn) {
    #pragma unroll 4
    for (int f = 0; f < FP; ++f) out[(size_t)f * NP] = (__bf16)0.0f;
    return;
  }
  const float* xr = inp + (((size_t)b * Ss + t) * Nn + n) * 21;
  float x2=xr[2], x5=xr[5], x8=xr[8], x9=xr[9], x12=xr[12];
  float x10=xr[10], x14=xr[14], x15=xr[15];
  float x13=xr[13], x17=xr[17], x19=xr[19], x20=xr[20];
  #pragma unroll
  for (int a = 0; a < 10; ++a) {
    float po = bo[a] + x2*Wo[a] + x5*Wo[10+a] + x8*Wo[20+a] + x9*Wo[30+a] + x12*Wo[40+a];
    float pd = bd[a] + x10*Wd[a] + x14*Wd[10+a] + x15*Wd[20+a];
    float pe = be[a] + x13*We[a] + x17*We[10+a];
    float pi = bi[a] + x19*Wi[a] + x20*Wi[10+a];
    out[(size_t)(a)      * NP] = (__bf16)po;
    out[(size_t)(10 + a) * NP] = (__bf16)pd;
    out[(size_t)(20 + a) * NP] = (__bf16)pe;
    out[(size_t)(30 + a) * NP] = (__bf16)pi;
  }
  const float* hr = h + ((size_t)b * Nn + n) * Hh;
  #pragma unroll 8
  for (int f = 0; f < Hh; ++f) out[(size_t)(40 + f) * NP] = (__bf16)hr[f];
  #pragma unroll
  for (int f = FCAT; f < FP; ++f) out[(size_t)f * NP] = (__bf16)0.0f;
}

// cat2: x part copied, hidden part = r (flattened-split gate) * h
__global__ void k_cat2(const __bf16* __restrict__ cat1T, const float* __restrict__ ru,
                       const float* __restrict__ h, __bf16* __restrict__ cat2T) {
  int idx = blockIdx.x * blockDim.x + threadIdx.x;
  int b = idx >> 10, n = idx & (NP - 1);
  __bf16* out = cat2T + ((size_t)b * FP) * NP + n;
  if (n >= Nn) {
    #pragma unroll 4
    for (int f = 0; f < FP; ++f) out[(size_t)f * NP] = (__bf16)0.0f;
    return;
  }
  const __bf16* in1 = cat1T + ((size_t)b * FP) * NP + n;
  #pragma unroll 8
  for (int f = 0; f < 40; ++f) out[(size_t)f * NP] = in1[(size_t)f * NP];
  const float* rub = ru + (size_t)b * Nn * G2;
  const float* hr  = h + ((size_t)b * Nn + n) * Hh;
  #pragma unroll 8
  for (int fh = 0; fh < Hh; ++fh) {
    int j = n * Hh + fh;                       // flat index into [N*H)
    float r = rub[(size_t)(j >> 7) * G2 + (j & 127)];  // r = first half of flat [N,2H]
    out[(size_t)(40 + fh) * NP] = (__bf16)(r * hr[fh]);
  }
  #pragma unroll
  for (int f = FCAT; f < FP; ++f) out[(size_t)f * NP] = (__bf16)0.0f;
}

// ---------------- fused gconv kernels (WMMA) ----------------
// Stage1: AG[16 x 112] = L[16 rows] @ cat_b[1024 x 112]   (bf16 wmma, f32 acc)
//   B (cat) K-chunks staged global->LDS (async, double buffered), shared by 4 waves.
// Stage2: gates = sigmoid(AG @ W1T + b1)   /  cand = tanh(AG @ W2T + b2) + GRU update
__global__ __launch_bounds__(128)
void k_gates(const __bf16* __restrict__ L, const __bf16* __restrict__ catT,
             const __bf16* __restrict__ W1T, const float* __restrict__ b1,
             float* __restrict__ ru) {
  const int b    = blockIdx.y;
  const int tid  = threadIdx.x;
  const int wave = tid >> 5;
  const int lane = tid & 31;
  const int mtile = blockIdx.x * 4 + wave;       // 0..63
  const int m0 = mtile * 16;
  const int row = lane & 15;
  const int hi  = (lane < 16) ? 0 : 1;

  __shared__ __align__(16) __bf16 bbuf[2][FP * BST];
  __shared__ __align__(16) __bf16 agl[4][16][LDSP];

  const __bf16* catB = catT + (size_t)b * FP * NP;
  const __bf16* Lrow = L + (size_t)(m0 + row) * NP + (hi ? 16 : 0);

  v8f acc[7];
  #pragma unroll
  for (int i = 0; i < 7; ++i) acc[i] = zero8();

  stage_chunk(catB, &bbuf[0][0], 0, tid);
  for (int kc = 0; kc < NKC; ++kc) {
    const int cur = kc & 1;
    if (kc + 1 < NKC) {
      stage_chunk(catB, &bbuf[cur ^ 1][0], kc + 1, tid);
      wait_async_outstanding7();       // chunk kc complete (own wave's ops)
    } else {
      wait_async_all();
    }
    __syncthreads();                   // chunk kc visible from all waves

    #pragma unroll
    for (int ks = 0; ks < 2; ++ks) {   // two K=32 steps per chunk
      const int k = kc * KC + ks * 32;
      v8bf alo = *(const v8bf*)(Lrow + k);
      v8bf ahi = *(const v8bf*)(Lrow + k + 8);
      v16bf a = cat16(alo, ahi);
      const int kbl = ks * 32 + hi * 16;
      #pragma unroll
      for (int ft = 0; ft < 7; ++ft) {
        const __bf16* bp = &bbuf[cur][(ft * 16 + row) * BST + kbl];
        v16bf bm = cat16(*(const v8bf*)bp, *(const v8bf*)(bp + 8));
        acc[ft] = __builtin_amdgcn_wmma_f32_16x16x32_bf16(
            false, a, false, bm, (short)0, acc[ft], false, false);
      }
    }
    __syncthreads();                   // done reading buf[cur] before it is reused
  }

  // spill AG -> LDS (bf16, A-layout source for stage 2)
  #pragma unroll
  for (int ft = 0; ft < 7; ++ft) {
    #pragma unroll
    for (int r = 0; r < 8; ++r)
      agl[wave][r + hi * 8][ft * 16 + row] = (__bf16)acc[ft][r];
  }
  { // zero pad K = 112..127
    int cbase = FP + hi * 8;
    #pragma unroll
    for (int c = 0; c < 8; ++c) agl[wave][row][cbase + c] = (__bf16)0.0f;
  }
  __syncthreads();

  v8f acc2[8];
  #pragma unroll
  for (int i = 0; i < 8; ++i) acc2[i] = zero8();
  #pragma unroll
  for (int k = 0; k < 128; k += 32) {
    const __bf16* ap = &agl[wave][row][k + (hi ? 16 : 0)];
    v16bf a = cat16(*(const v8bf*)ap, *(const v8bf*)(ap + 8));
    const int kb = k + (hi ? 16 : 0);
    #pragma unroll
    for (int ot = 0; ot < 8; ++ot) {
      const __bf16* bp = W1T + (size_t)(ot * 16 + row) * 128 + kb;
      v16bf bm = cat16(*(const v8bf*)bp, *(const v8bf*)(bp + 8));
      acc2[ot] = __builtin_amdgcn_wmma_f32_16x16x32_bf16(
          false, a, false, bm, (short)0, acc2[ot], false, false);
    }
  }

  float* rub = ru + (size_t)b * Nn * G2;
  #pragma unroll
  for (int ot = 0; ot < 8; ++ot) {
    int col = ot * 16 + row;
    float bv = b1[col];
    #pragma unroll
    for (int r = 0; r < 8; ++r) {
      int m = m0 + r + hi * 8;
      if (m < Nn) {
        float z = acc2[ot][r] + bv;
        rub[(size_t)m * G2 + col] = 1.0f / (1.0f + expf(-z));
      }
    }
  }
}

__global__ __launch_bounds__(128)
void k_cand(const __bf16* __restrict__ L, const __bf16* __restrict__ catT,
            const __bf16* __restrict__ W2T, const float* __restrict__ b2,
            const float* __restrict__ ru, float* __restrict__ h) {
  const int b    = blockIdx.y;
  const int tid  = threadIdx.x;
  const int wave = tid >> 5;
  const int lane = tid & 31;
  const int mtile = blockIdx.x * 4 + wave;
  const int m0 = mtile * 16;
  const int row = lane & 15;
  const int hi  = (lane < 16) ? 0 : 1;

  __shared__ __align__(16) __bf16 bbuf[2][FP * BST];
  __shared__ __align__(16) __bf16 agl[4][16][LDSP];

  const __bf16* catB = catT + (size_t)b * FP * NP;
  const __bf16* Lrow = L + (size_t)(m0 + row) * NP + (hi ? 16 : 0);

  v8f acc[7];
  #pragma unroll
  for (int i = 0; i < 7; ++i) acc[i] = zero8();

  stage_chunk(catB, &bbuf[0][0], 0, tid);
  for (int kc = 0; kc < NKC; ++kc) {
    const int cur = kc & 1;
    if (kc + 1 < NKC) {
      stage_chunk(catB, &bbuf[cur ^ 1][0], kc + 1, tid);
      wait_async_outstanding7();
    } else {
      wait_async_all();
    }
    __syncthreads();

    #pragma unroll
    for (int ks = 0; ks < 2; ++ks) {
      const int k = kc * KC + ks * 32;
      v8bf alo = *(const v8bf*)(Lrow + k);
      v8bf ahi = *(const v8bf*)(Lrow + k + 8);
      v16bf a = cat16(alo, ahi);
      const int kbl = ks * 32 + hi * 16;
      #pragma unroll
      for (int ft = 0; ft < 7; ++ft) {
        const __bf16* bp = &bbuf[cur][(ft * 16 + row) * BST + kbl];
        v16bf bm = cat16(*(const v8bf*)bp, *(const v8bf*)(bp + 8));
        acc[ft] = __builtin_amdgcn_wmma_f32_16x16x32_bf16(
            false, a, false, bm, (short)0, acc[ft], false, false);
      }
    }
    __syncthreads();
  }

  #pragma unroll
  for (int ft = 0; ft < 7; ++ft) {
    #pragma unroll
    for (int r = 0; r < 8; ++r)
      agl[wave][r + hi * 8][ft * 16 + row] = (__bf16)acc[ft][r];
  }
  {
    int cbase = FP + hi * 8;
    #pragma unroll
    for (int c = 0; c < 8; ++c) agl[wave][row][cbase + c] = (__bf16)0.0f;
  }
  __syncthreads();

  v8f acc2[4];
  #pragma unroll
  for (int i = 0; i < 4; ++i) acc2[i] = zero8();
  #pragma unroll
  for (int k = 0; k < 128; k += 32) {
    const __bf16* ap = &agl[wave][row][k + (hi ? 16 : 0)];
    v16bf a = cat16(*(const v8bf*)ap, *(const v8bf*)(ap + 8));
    const int kb = k + (hi ? 16 : 0);
    #pragma unroll
    for (int ot = 0; ot < 4; ++ot) {
      const __bf16* bp = W2T + (size_t)(ot * 16 + row) * 128 + kb;
      v16bf bm = cat16(*(const v8bf*)bp, *(const v8bf*)(bp + 8));
      acc2[ot] = __builtin_amdgcn_wmma_f32_16x16x32_bf16(
          false, a, false, bm, (short)0, acc2[ot], false, false);
    }
  }

  const float* rub = ru + (size_t)b * Nn * G2;
  float* hb = h + (size_t)b * Nn * Hh;
  #pragma unroll
  for (int ot = 0; ot < 4; ++ot) {
    int col = ot * 16 + row;        // 0..63
    float bv = b2[col];
    #pragma unroll
    for (int r = 0; r < 8; ++r) {
      int m = m0 + r + hi * 8;
      if (m < Nn) {
        float c = tanhf(acc2[ot][r] + bv);
        int j = m * Hh + col;       // flat [N*H) index
        float u = rub[(size_t)(500 + (j >> 7)) * G2 + (j & 127)]; // u = 2nd half of flat split
        float hold = hb[(size_t)m * Hh + col];
        hb[(size_t)m * Hh + col] = u * hold + (1.0f - u) * c;
      }
    }
  }
}

// ---------------- launch ----------------
extern "C" void kernel_launch(void* const* d_in, const int* in_sizes, int n_in,
                              void* d_out, int out_size, void* d_ws, size_t ws_size,
                              hipStream_t stream) {
  const float* inp = (const float*)d_in[0];
  const float* adj = (const float*)d_in[1];
  const float* Wo  = (const float*)d_in[2];
  const float* bo  = (const float*)d_in[3];
  const float* Wd  = (const float*)d_in[4];
  const float* bd  = (const float*)d_in[5];
  const float* We  = (const float*)d_in[6];
  const float* be  = (const float*)d_in[7];
  const float* Wi  = (const float*)d_in[8];
  const float* bi  = (const float*)d_in[9];
  const float* W1  = (const float*)d_in[10];
  const float* b1  = (const float*)d_in[11];
  const float* W2  = (const float*)d_in[12];
  const float* b2  = (const float*)d_in[13];
  float* h = (float*)d_out;                     // [B][N][H], doubles as GRU state

  char* ws = (char*)d_ws;
  float*  dinv  = (float*)(ws + 0);                      //   4 KB
  __bf16* Lbf   = (__bf16*)(ws + 4096);                  //   2 MB
  __bf16* W1T   = (__bf16*)(ws + 4096 + 2097152);        //  32 KB
  __bf16* W2T   = (__bf16*)(ws + 4096 + 2097152 + 32768);//  16 KB
  __bf16* cat1T = (__bf16*)(ws + 2150400);               // 7.34 MB
  __bf16* cat2T = (__bf16*)(ws + 9490432);               // 7.34 MB
  float*  ru    = (float*)(ws + 16830464);               // 16.4 MB

  // setup: Laplacian (bf16, padded) + transposed bf16 weights + h0 = 0
  k_dinv<<<4, 256, 0, stream>>>(adj, dinv);
  k_buildL<<<(NP * NP) / 256, 256, 0, stream>>>(adj, dinv, Lbf);
  k_buildW<<<(G2 * 128 + Hh * 128) / 256, 256, 0, stream>>>(W1, W2, W1T, W2T);
  (void)hipMemsetAsync(h, 0, (size_t)Bb * Nn * Hh * sizeof(float), stream);

  dim3 gW(16, Bb);   // 16 blocks * 4 waves = 64 m-tiles, per batch
  for (int t = 0; t < Ss; ++t) {
    k_cat1<<<(Bb * NP) / 256, 256, 0, stream>>>(inp, t, Wo, bo, Wd, bd, We, be,
                                                Wi, bi, h, cat1T);
    k_gates<<<gW, 128, 0, stream>>>(Lbf, cat1T, W1T, b1, ru);
    k_cat2<<<(Bb * NP) / 256, 256, 0, stream>>>(cat1T, ru, h, cat2T);
    k_cand<<<gW, 128, 0, stream>>>(Lbf, cat2T, W2T, b2, ru, h);
  }
}
